// CorticalColumnBlock_62182536512334
// MI455X (gfx1250) — matrix operand order repaired and checked by reading, not compiled, not run on hardware
//
#include <hip/hip_runtime.h>
#include <hip/hip_bf16.h>
#include <math.h>

typedef _Float16 h16;
typedef __attribute__((ext_vector_type(16))) _Float16 v16h;
typedef __attribute__((ext_vector_type(32))) _Float16 v32h;
typedef __attribute__((ext_vector_type(8)))  _Float16 v8h;
typedef __attribute__((ext_vector_type(8)))  float    v8f;

#define B_  4
#define T_  1024
#define C_  1024
#define BT_ 4096
#define H_  16
#define HD_ 64
#define NC_ 16
#define CS_ 64

union Frag  { v16h v; v8h h8[2]; h16 e[16]; };
union FragB { v32h v; v8h h8[4]; h16 e[32]; };

__device__ __forceinline__ float gelu_exact(float x) {
    return 0.5f * x * (1.0f + erff(x * 0.70710678118654752f));
}

// ---------------------------------------------------------------------------
// Weight conversion: dst = (h16)(src * mask?) elementwise
// ---------------------------------------------------------------------------
__global__ void cvt_kernel(const float* __restrict__ src, const float* __restrict__ mask,
                           h16* __restrict__ dst, int n) {
    int i = blockIdx.x * blockDim.x + threadIdx.x;
    int stride = gridDim.x * blockDim.x;
    for (; i < n; i += stride) {
        float v = src[i];
        if (mask) v *= mask[i];
        dst[i] = (h16)v;
    }
}

// ---------------------------------------------------------------------------
// Build 2:4 sparse index words from mask2 (column pattern shared by all rows).
// idxw[w] covers original K columns [32w, 32w+32): 8 groups x 4 bits
// (bits[1:0]=idx0, bits[3:2]=idx1 per the SWMMAC index layout).
// ---------------------------------------------------------------------------
__global__ void w2_idx_kernel(const float* __restrict__ mask2, unsigned* __restrict__ idxw) {
    int w = blockIdx.x * blockDim.x + threadIdx.x;
    if (w >= (4 * C_) / 32) return;
    unsigned word = 0;
    for (int j = 0; j < 8; ++j) {
        int gi = w * 8 + j;
        int i0 = -1, i1 = -1;
        for (int t = 0; t < 4; ++t)
            if (mask2[4 * gi + t] != 0.0f) { if (i0 < 0) i0 = t; else if (i1 < 0) i1 = t; }
        if (i0 < 0) i0 = 0;
        if (i1 < 0) i1 = (i0 == 0) ? 1 : 0;
        word |= (unsigned)((i0 & 3) | ((i1 & 3) << 2)) << (4 * j);
    }
    idxw[w] = word;
}

// ---------------------------------------------------------------------------
// Pack w2*mask2 into half-size 2:4 A-operand: w2p[row][2*gi+{0,1}] = kept pair
// ---------------------------------------------------------------------------
__global__ void w2_pack_kernel(const float* __restrict__ w2, const float* __restrict__ mask2,
                               h16* __restrict__ w2p) {
    int idx = blockIdx.x * blockDim.x + threadIdx.x;
    int total = C_ * 1024;            // rows * K-groups
    int stride = gridDim.x * blockDim.x;
    for (; idx < total; idx += stride) {
        int row = idx >> 10;
        int gi  = idx & 1023;
        int i0 = -1, i1 = -1;
        for (int t = 0; t < 4; ++t)
            if (mask2[4 * gi + t] != 0.0f) { if (i0 < 0) i0 = t; else if (i1 < 0) i1 = t; }
        if (i0 < 0) i0 = 0;
        if (i1 < 0) i1 = (i0 == 0) ? 1 : 0;
        const float* wr = w2 + (size_t)row * (4 * C_) + 4 * gi;
        const float* mr = mask2 + 4 * gi;
        w2p[(size_t)row * (2 * C_) + 2 * gi + 0] = (h16)(wr[i0] * mr[i0]);
        w2p[(size_t)row * (2 * C_) + 2 * gi + 1] = (h16)(wr[i1] * mr[i1]);
    }
}

// ---------------------------------------------------------------------------
// LN1 + gate (sigmoid(gelu(h1 @ gw1^T) @ gw2^T) > 0.5) -> mx = h1*imp (f16)
// ---------------------------------------------------------------------------
__global__ void ln1_gate_kernel(const float* __restrict__ x,
                                const float* __restrict__ lnw, const float* __restrict__ lnb,
                                const float* __restrict__ gw1, const float* __restrict__ gw2,
                                h16* __restrict__ mxh, float* __restrict__ imp) {
    __shared__ float red[256];
    __shared__ float h1s[C_];
    __shared__ float ys[128];
    int row = blockIdx.x;
    int tid = threadIdx.x;
    const float* xr = x + (size_t)row * C_;

    float s = 0.f;
    for (int c = tid; c < C_; c += 256) s += xr[c];
    red[tid] = s; __syncthreads();
    for (int o = 128; o > 0; o >>= 1) { if (tid < o) red[tid] += red[tid + o]; __syncthreads(); }
    float mu = red[0] * (1.0f / C_);
    __syncthreads();

    s = 0.f;
    for (int c = tid; c < C_; c += 256) { float d = xr[c] - mu; s += d * d; }
    red[tid] = s; __syncthreads();
    for (int o = 128; o > 0; o >>= 1) { if (tid < o) red[tid] += red[tid + o]; __syncthreads(); }
    float rstd = rsqrtf(red[0] * (1.0f / C_) + 1e-5f);
    __syncthreads();

    for (int c = tid; c < C_; c += 256)
        h1s[c] = (xr[c] - mu) * rstd * lnw[c] + lnb[c];
    __syncthreads();

    if (tid < 128) {
        const float* wr = gw1 + (size_t)tid * C_;
        float acc = 0.f;
        for (int c = 0; c < C_; ++c) acc += h1s[c] * wr[c];
        ys[tid] = gelu_exact(acc) * gw2[tid];
    }
    __syncthreads();
    if (tid < 64) red[tid] = ys[tid] + ys[tid + 64];
    __syncthreads();
    for (int o = 32; o > 0; o >>= 1) { if (tid < o) red[tid] += red[tid + o]; __syncthreads(); }
    float g  = 1.0f / (1.0f + __expf(-red[0]));
    float im = (g > 0.5f) ? 1.0f : 0.0f;
    if (tid == 0) imp[row] = im;
    for (int c = tid; c < C_; c += 256)
        mxh[(size_t)row * C_ + c] = (h16)(h1s[c] * im);
}

// ---------------------------------------------------------------------------
// LN2: x2 -> h2 (f16)
// ---------------------------------------------------------------------------
__global__ void ln2_kernel(const float* __restrict__ x,
                           const float* __restrict__ lnw, const float* __restrict__ lnb,
                           h16* __restrict__ out) {
    __shared__ float red[256];
    int row = blockIdx.x;
    int tid = threadIdx.x;
    const float* xr = x + (size_t)row * C_;

    float s = 0.f;
    for (int c = tid; c < C_; c += 256) s += xr[c];
    red[tid] = s; __syncthreads();
    for (int o = 128; o > 0; o >>= 1) { if (tid < o) red[tid] += red[tid + o]; __syncthreads(); }
    float mu = red[0] * (1.0f / C_);
    __syncthreads();
    s = 0.f;
    for (int c = tid; c < C_; c += 256) { float d = xr[c] - mu; s += d * d; }
    red[tid] = s; __syncthreads();
    for (int o = 128; o > 0; o >>= 1) { if (tid < o) red[tid] += red[tid + o]; __syncthreads(); }
    float rstd = rsqrtf(red[0] * (1.0f / C_) + 1e-5f);
    for (int c = tid; c < C_; c += 256)
        out[(size_t)row * C_ + c] = (h16)((xr[c] - mu) * rstd * lnw[c] + lnb[c]);
}

// ---------------------------------------------------------------------------
// Wave-level GEMM core: one wave computes 16 (M) x 64 (N) f32 tile.
// ---------------------------------------------------------------------------
template <int K>
__device__ __forceinline__ void wave_gemm_16x64(const h16* __restrict__ arow,
                                                const h16* __restrict__ W, int n0,
                                                int g, int ln, v8f acc[4]) {
    for (int k0 = 0; k0 < K; k0 += 32) {
        Frag a;
        a.h8[0] = *(const v8h*)(arow + k0 + g * 8);
        a.h8[1] = *(const v8h*)(arow + k0 + 16 + g * 8);
#pragma unroll
        for (int nt = 0; nt < 4; ++nt) {
            Frag b;
            const h16* wp = W + (size_t)(n0 + nt * 16 + ln) * K + k0 + g * 8;
            b.h8[0] = *(const v8h*)(wp);
            b.h8[1] = *(const v8h*)(wp + 16);
            acc[nt] = __builtin_amdgcn_wmma_f32_16x16x32_f16(
                false, a.v, false, b.v, (short)0, acc[nt], false, false);
        }
    }
}

// ---------------------------------------------------------------------------
// QKV GEMM: mx[BT,C] x Wqkv[3072,C]^T -> q,k:[B,H,T,HD], v:[B,H,HD,T] (f16)
// ---------------------------------------------------------------------------
__global__ void qkv_gemm_kernel(const h16* __restrict__ A, const h16* __restrict__ W,
                                h16* __restrict__ qh, h16* __restrict__ kh,
                                h16* __restrict__ vT) {
    int lane = threadIdx.x & 31;
    int g = lane >> 4, ln = lane & 15;
    int n0 = blockIdx.x * 64;
    int m0 = blockIdx.y * 16;
    v8f acc[4] = {v8f{}, v8f{}, v8f{}, v8f{}};
    const h16* arow = A + (size_t)(m0 + ln) * C_;
    wave_gemm_16x64<C_>(arow, W, n0, g, ln, acc);

#pragma unroll
    for (int nt = 0; nt < 4; ++nt) {
        int o   = n0 + nt * 16;
        int d   = o / 768;
        int rem = o % 768;
        int seg = rem / 256;
        int oo  = rem % 256;
        int h   = d * 4 + oo / 64;
        int e   = (oo % 64) + ln;
#pragma unroll
        for (int r = 0; r < 8; ++r) {
            int m  = m0 + r + 8 * g;
            int bb = m >> 10;
            int tt = m & 1023;
            h16 val = (h16)acc[nt][r];
            if (seg == 0)
                qh[((size_t)(bb * H_ + h) * T_ + tt) * HD_ + e] = val;
            else if (seg == 1)
                kh[((size_t)(bb * H_ + h) * T_ + tt) * HD_ + e] = val;
            else
                vT[((size_t)(bb * H_ + h) * HD_ + e) * T_ + tt] = val;
        }
    }
}

// ---------------------------------------------------------------------------
// Flash attention: one wave per (b,h,16-query tile). Causal, HD=64.
// ---------------------------------------------------------------------------
__global__ void attn_kernel(const h16* __restrict__ qh, const h16* __restrict__ kh,
                            const h16* __restrict__ vT, h16* __restrict__ attout) {
    __shared__ h16 pbuf[16 * 32];
    int lane = threadIdx.x & 31;
    int g = lane >> 4, ln = lane & 15;
    int q0 = blockIdx.x * 16;
    int h  = blockIdx.y;
    int b  = blockIdx.z;
    const h16* qbase = qh + (size_t)(b * H_ + h) * T_ * HD_;
    const h16* kbase = kh + (size_t)(b * H_ + h) * T_ * HD_;
    const h16* vbase = vT + (size_t)(b * H_ + h) * HD_ * T_;

    Frag qa0, qa1;
    {
        const h16* qr = qbase + (size_t)(q0 + ln) * HD_;
        qa0.h8[0] = *(const v8h*)(qr + g * 8);
        qa0.h8[1] = *(const v8h*)(qr + 16 + g * 8);
        qa1.h8[0] = *(const v8h*)(qr + 32 + g * 8);
        qa1.h8[1] = *(const v8h*)(qr + 48 + g * 8);
    }

    v8f o[4] = {v8f{}, v8f{}, v8f{}, v8f{}};
    float rmax[8], rsum[8];
#pragma unroll
    for (int r = 0; r < 8; ++r) { rmax[r] = -__builtin_inff(); rsum[r] = 0.f; }

    for (int k0 = 0; k0 <= q0 + 15; k0 += 32) {
#pragma unroll
        for (int sub = 0; sub < 2; ++sub) {
            int kb = k0 + sub * 16;
            Frag kf0, kf1;
            const h16* kr = kbase + (size_t)(kb + ln) * HD_;
            kf0.h8[0] = *(const v8h*)(kr + g * 8);
            kf0.h8[1] = *(const v8h*)(kr + 16 + g * 8);
            kf1.h8[0] = *(const v8h*)(kr + 32 + g * 8);
            kf1.h8[1] = *(const v8h*)(kr + 48 + g * 8);
            v8f s = {};
            s = __builtin_amdgcn_wmma_f32_16x16x32_f16(false, qa0.v, false, kf0.v,
                                                       (short)0, s, false, false);
            s = __builtin_amdgcn_wmma_f32_16x16x32_f16(false, qa1.v, false, kf1.v,
                                                       (short)0, s, false, false);
#pragma unroll
            for (int r = 0; r < 8; ++r) {
                int qrow = q0 + r + 8 * g;
                int key  = kb + ln;
                float sv = (key <= qrow) ? s[r] * 0.125f : -__builtin_inff();
                float tmax = sv;
                for (int msk = 1; msk < 16; msk <<= 1)
                    tmax = fmaxf(tmax, __shfl_xor(tmax, msk, 32));
                float nmax = fmaxf(rmax[r], tmax);
                float p, factor;
                if (nmax == -__builtin_inff()) { factor = 1.f; p = 0.f; }
                else {
                    factor = __expf(rmax[r] - nmax);
                    p = (sv == -__builtin_inff()) ? 0.f : __expf(sv - nmax);
                }
                float psum = p;
                for (int msk = 1; msk < 16; msk <<= 1)
                    psum += __shfl_xor(psum, msk, 32);
                rsum[r] = rsum[r] * factor + psum;
                rmax[r] = nmax;
                o[0][r] *= factor; o[1][r] *= factor; o[2][r] *= factor; o[3][r] *= factor;
                pbuf[(r + 8 * g) * 32 + sub * 16 + ln] = (h16)p;
            }
        }
        __syncthreads();
        Frag pa;
        pa.h8[0] = *(const v8h*)(&pbuf[ln * 32 + g * 8]);
        pa.h8[1] = *(const v8h*)(&pbuf[ln * 32 + 16 + g * 8]);
#pragma unroll
        for (int nt = 0; nt < 4; ++nt) {
            Frag vb;
            const h16* vr = vbase + (size_t)(nt * 16 + ln) * T_ + k0 + g * 8;
            vb.h8[0] = *(const v8h*)(vr);
            vb.h8[1] = *(const v8h*)(vr + 16);
            o[nt] = __builtin_amdgcn_wmma_f32_16x16x32_f16(
                false, pa.v, false, vb.v, (short)0, o[nt], false, false);
        }
        __syncthreads();
    }

#pragma unroll
    for (int r = 0; r < 8; ++r) {
        int qrow  = q0 + r + 8 * g;
        float inv = 1.0f / rsum[r];
        size_t base = ((size_t)b * T_ + qrow) * C_ + h * HD_;
#pragma unroll
        for (int nt = 0; nt < 4; ++nt)
            attout[base + nt * 16 + ln] = (h16)(o[nt][r] * inv);
    }
}

// ---------------------------------------------------------------------------
// Soma projection + gate + residual: x2 = x + (attout @ soma^T) * imp
// ---------------------------------------------------------------------------
__global__ void soma_gemm_kernel(const h16* __restrict__ A, const h16* __restrict__ W,
                                 const float* __restrict__ x, const float* __restrict__ imp,
                                 float* __restrict__ x2) {
    int lane = threadIdx.x & 31;
    int g = lane >> 4, ln = lane & 15;
    int n0 = blockIdx.x * 64;
    int m0 = blockIdx.y * 16;
    v8f acc[4] = {v8f{}, v8f{}, v8f{}, v8f{}};
    wave_gemm_16x64<C_>(A + (size_t)(m0 + ln) * C_, W, n0, g, ln, acc);
#pragma unroll
    for (int nt = 0; nt < 4; ++nt)
#pragma unroll
        for (int r = 0; r < 8; ++r) {
            int m = m0 + r + 8 * g;
            size_t idx = (size_t)m * C_ + n0 + nt * 16 + ln;
            x2[idx] = x[idx] + acc[nt][r] * imp[m];
        }
}

// ---------------------------------------------------------------------------
// MLP1: hmid = gelu(h2 @ (w1*mask1)^T)
// ---------------------------------------------------------------------------
__global__ void mlp1_gemm_kernel(const h16* __restrict__ A, const h16* __restrict__ W,
                                 h16* __restrict__ out) {
    int lane = threadIdx.x & 31;
    int g = lane >> 4, ln = lane & 15;
    int n0 = blockIdx.x * 64;
    int m0 = blockIdx.y * 16;
    v8f acc[4] = {v8f{}, v8f{}, v8f{}, v8f{}};
    wave_gemm_16x64<C_>(A + (size_t)(m0 + ln) * C_, W, n0, g, ln, acc);
#pragma unroll
    for (int nt = 0; nt < 4; ++nt)
#pragma unroll
        for (int r = 0; r < 8; ++r) {
            int m = m0 + r + 8 * g;
            out[(size_t)m * (4 * C_) + n0 + nt * 16 + ln] = (h16)gelu_exact(acc[nt][r]);
        }
}

// ---------------------------------------------------------------------------
// MLP2 via 2:4 structured-sparse SWMMAC (A = w2 rows, sparse along K=4C;
// B = h^T, dense). Computes out^T tiles: M=16 out-cols, N=4x16 bt rows.
// grid (C/16, BT/64), block 32
// ---------------------------------------------------------------------------
#if __has_builtin(__builtin_amdgcn_swmmac_f32_16x16x64_f16)
#define HAVE_SWMMAC 1
__global__ void mlp2_swmmac_kernel(const h16* __restrict__ Hm, const h16* __restrict__ W2p,
                                   const unsigned* __restrict__ idxw, float* __restrict__ out) {
    int lane = threadIdx.x & 31;
    int g = lane >> 4, ln = lane & 15;
    int n0 = blockIdx.x * 16;   // out-column base (rows of w2)
    int m0 = blockIdx.y * 64;   // bt-row base
    v8f acc[4] = {v8f{}, v8f{}, v8f{}, v8f{}};
    const h16* arow = W2p + (size_t)(n0 + ln) * (2 * C_);
    for (int k0 = 0; k0 < 4 * C_; k0 += 64) {
        Frag a;                                   // packed 16x32 A operand
        int kp = k0 >> 1;
        a.h8[0] = *(const v8h*)(arow + kp + g * 8);
        a.h8[1] = *(const v8h*)(arow + kp + 16 + g * 8);
        int sidx = (int)idxw[(k0 >> 5) + g];      // lane group g: cols g*32..g*32+31
#pragma unroll
        for (int nt = 0; nt < 4; ++nt) {
            FragB b;                              // dense 64x16 B operand
            const h16* hr = Hm + (size_t)(m0 + nt * 16 + ln) * (4 * C_) + k0;
            b.h8[0] = *(const v8h*)(hr + g * 16);
            b.h8[1] = *(const v8h*)(hr + g * 16 + 8);
            b.h8[2] = *(const v8h*)(hr + 32 + g * 16);
            b.h8[3] = *(const v8h*)(hr + 32 + g * 16 + 8);
            acc[nt] = __builtin_amdgcn_swmmac_f32_16x16x64_f16(
                false, a.v, false, b.v, acc[nt], sidx, false, false);
        }
    }
#pragma unroll
    for (int nt = 0; nt < 4; ++nt)
#pragma unroll
        for (int r = 0; r < 8; ++r)
            out[(size_t)(m0 + nt * 16 + ln) * C_ + n0 + r + 8 * g] = acc[nt][r];
}
#else
#define HAVE_SWMMAC 0
#endif

// Dense fallback MLP2 (also used if SWMMAC builtin absent)
__global__ void mlp2_gemm_kernel(const h16* __restrict__ A, const h16* __restrict__ W,
                                 float* __restrict__ out) {
    int lane = threadIdx.x & 31;
    int g = lane >> 4, ln = lane & 15;
    int n0 = blockIdx.x * 64;
    int m0 = blockIdx.y * 16;
    v8f acc[4] = {v8f{}, v8f{}, v8f{}, v8f{}};
    wave_gemm_16x64<4 * C_>(A + (size_t)(m0 + ln) * (4 * C_), W, n0, g, ln, acc);
#pragma unroll
    for (int nt = 0; nt < 4; ++nt)
#pragma unroll
        for (int r = 0; r < 8; ++r) {
            int m = m0 + r + 8 * g;
            out[(size_t)m * C_ + n0 + nt * 16 + ln] = acc[nt][r];
        }
}

// ---------------------------------------------------------------------------
// Lateral inhibition + residual: out = x2 + relu(m * (1 - inh[col]))
// ---------------------------------------------------------------------------
__global__ void inhib_kernel(const float* __restrict__ mlp, const float* __restrict__ x2,
                             const float* __restrict__ inhib, float* __restrict__ out) {
    __shared__ float act[NC_];
    __shared__ float inh[NC_];
    int row = blockIdx.x, tid = threadIdx.x;
    const float* mr = mlp + (size_t)row * C_;
    if (tid < NC_) {
        float s = 0.f;
        for (int j = 0; j < CS_; ++j) s += mr[tid * CS_ + j];
        act[tid] = s * (1.0f / CS_);
    }
    __syncthreads();
    if (tid < NC_) {
        float s = 0.f;
        for (int j = 0; j < NC_; ++j) s += act[j] * inhib[j * NC_ + tid];
        inh[tid] = s;
    }
    __syncthreads();
    for (int c = tid; c < C_; c += 256) {
        float m = mr[c];
        float v = m - inh[c >> 6] * m;
        size_t idx = (size_t)row * C_ + c;
        out[idx] = x2[idx] + fmaxf(v, 0.f);
    }
}

// ---------------------------------------------------------------------------
extern "C" void kernel_launch(void* const* d_in, const int* in_sizes, int n_in,
                              void* d_out, int out_size, void* d_ws, size_t ws_size,
                              hipStream_t stream) {
    const float* x      = (const float*)d_in[0];
    const float* ln1w   = (const float*)d_in[1];
    const float* ln1b   = (const float*)d_in[2];
    const float* ln2w   = (const float*)d_in[3];
    const float* ln2b   = (const float*)d_in[4];
    const float* qkv_w  = (const float*)d_in[5];
    const float* soma_w = (const float*)d_in[6];
    const float* gw1    = (const float*)d_in[7];
    const float* gw2    = (const float*)d_in[8];
    const float* w1     = (const float*)d_in[9];
    const float* w2     = (const float*)d_in[10];
    const float* mask1  = (const float*)d_in[11];
    const float* mask2  = (const float*)d_in[12];
    const float* inhib  = (const float*)d_in[13];

    char*  base = (char*)d_ws;
    size_t off  = 0;
    auto alloc = [&](size_t bytes) -> void* {
        void* p = base + off;
        off = (off + bytes + 255) & ~(size_t)255;
        return p;
    };
    h16*      mxh   = (h16*)alloc((size_t)BT_ * C_ * 2);
    float*    impv  = (float*)alloc((size_t)BT_ * 4);
    h16*      qh    = (h16*)alloc((size_t)B_ * H_ * T_ * HD_ * 2);
    h16*      kh    = (h16*)alloc((size_t)B_ * H_ * T_ * HD_ * 2);
    h16*      vTh   = (h16*)alloc((size_t)B_ * H_ * HD_ * T_ * 2);
    h16*      atth  = (h16*)alloc((size_t)BT_ * C_ * 2);
    float*    x2    = (float*)alloc((size_t)BT_ * C_ * 4);
    h16*      h2h   = (h16*)alloc((size_t)BT_ * C_ * 2);
    h16*      hmid  = (h16*)alloc((size_t)BT_ * 4 * C_ * 2);
    float*    mlpo  = (float*)alloc((size_t)BT_ * C_ * 4);
    h16*      wqkvh = (h16*)alloc((size_t)3072 * 1024 * 2);
    h16*      wsomh = (h16*)alloc((size_t)1024 * 1024 * 2);
    h16*      w1mh  = (h16*)alloc((size_t)4096 * 1024 * 2);
#if HAVE_SWMMAC
    h16*      w2p   = (h16*)alloc((size_t)C_ * (2 * C_) * 2);
    unsigned* idxw  = (unsigned*)alloc((size_t)((4 * C_) / 32) * 4);
#else
    h16*      w2mh  = (h16*)alloc((size_t)1024 * 4096 * 2);
#endif

    cvt_kernel<<<512, 256, 0, stream>>>(qkv_w, nullptr, wqkvh, 3072 * 1024);
    cvt_kernel<<<512, 256, 0, stream>>>(soma_w, nullptr, wsomh, 1024 * 1024);
    cvt_kernel<<<512, 256, 0, stream>>>(w1, mask1, w1mh, 4096 * 1024);
#if HAVE_SWMMAC
    w2_idx_kernel<<<1, 128, 0, stream>>>(mask2, idxw);
    w2_pack_kernel<<<1024, 256, 0, stream>>>(w2, mask2, w2p);
#else
    cvt_kernel<<<512, 256, 0, stream>>>(w2, mask2, w2mh, 1024 * 4096);
#endif

    ln1_gate_kernel<<<BT_, 256, 0, stream>>>(x, ln1w, ln1b, gw1, gw2, mxh, impv);
    qkv_gemm_kernel<<<dim3(3072 / 64, BT_ / 16), 32, 0, stream>>>(mxh, wqkvh, qh, kh, vTh);
    attn_kernel<<<dim3(T_ / 16, H_, B_), 32, 0, stream>>>(qh, kh, vTh, atth);
    soma_gemm_kernel<<<dim3(C_ / 64, BT_ / 16), 32, 0, stream>>>(atth, wsomh, x, impv, x2);
    ln2_kernel<<<BT_, 256, 0, stream>>>(x2, ln2w, ln2b, h2h);
    mlp1_gemm_kernel<<<dim3(4 * C_ / 64, BT_ / 16), 32, 0, stream>>>(h2h, w1mh, hmid);
#if HAVE_SWMMAC
    mlp2_swmmac_kernel<<<dim3(C_ / 16, BT_ / 64), 32, 0, stream>>>(hmid, w2p, idxw, mlpo);
#else
    mlp2_gemm_kernel<<<dim3(C_ / 64, BT_ / 16), 32, 0, stream>>>(hmid, w2mh, mlpo);
#endif
    inhib_kernel<<<BT_, 256, 0, stream>>>(mlpo, x2, inhib, (float*)d_out);
}
